// SwinTransformerBlock_82265803587932
// MI455X (gfx1250) — matrix operand order repaired and checked by reading, not compiled, not run on hardware
//
#include <hip/hip_runtime.h>
#include <hip/hip_bf16.h>
#include <math.h>

// ---------------------------------------------------------------------------
// Types for CDNA5 WMMA (wave32): V_WMMA_F32_16X16X32_BF16
// ---------------------------------------------------------------------------
typedef __bf16 bf16_t;
typedef __attribute__((ext_vector_type(8)))  __bf16 v8bf;
typedef __attribute__((ext_vector_type(16))) __bf16 v16bf;
typedef __attribute__((ext_vector_type(8)))  float  v8f;

static __device__ __forceinline__ v8f wmma_bf16(v16bf a, v16bf b, v8f c) {
  // (neg_a, A, neg_b, B, c_mod, C, reuse_a, reuse_b)
  return __builtin_amdgcn_wmma_f32_16x16x32_bf16(false, a, false, b, (short)0, c,
                                                 false, false);
}

// B-operand: lane (l&15) holds column, elements 0..15 = contiguous K strip.
static __device__ __forceinline__ v16bf load_contig16(const bf16_t* p) {
  v8bf lo = *(const v8bf*)p;
  v8bf hi = *(const v8bf*)(p + 8);
  v16bf r;
#pragma unroll
  for (int i = 0; i < 8; ++i) { r[i] = lo[i]; r[8 + i] = hi[i]; }
  return r;
}

// A-operand: per-lane elements 0..15 = contiguous K strip at p (K+0..15).
// (lane>>4)*16 already folded into p by the caller.
static __device__ __forceinline__ v16bf load_a16(const bf16_t* p) {
  v8bf lo = *(const v8bf*)p;
  v8bf hi = *(const v8bf*)(p + 8);
  v16bf r;
#pragma unroll
  for (int i = 0; i < 8; ++i) { r[i] = lo[i]; r[8 + i] = hi[i]; }
  return r;
}

// ---------------------------------------------------------------------------
// Problem constants
// ---------------------------------------------------------------------------
#define BB   4
#define HH   64
#define WW   64
#define CC   768
#define WS_  14
#define NH_  12
#define HD_  64
#define NWIN 5                    // per side (70/14)
#define BW   (BB * NWIN * NWIN)   // 100 windows
#define NT   (WS_ * WS_)          // 196 tokens / window
#define NPAIR (BW * NH_)          // 1200 (window, head) pairs
#define MQ   (BW * NT)            // 19600 rows in window space
#define MT   (BB * HH * WW)       // 16384 tokens
#define VPAD 224                  // padded key dim for V^T (7 * 32)

// ---------------------------------------------------------------------------
// Weight fp32 [K,N] -> bf16 [N,K] transpose
// ---------------------------------------------------------------------------
__global__ __launch_bounds__(256) void k_transpose_w(const float* __restrict__ w,
                                                     bf16_t* __restrict__ wT,
                                                     int K, int N) {
  size_t i = (size_t)blockIdx.x * 256 + threadIdx.x;
  if (i >= (size_t)K * N) return;
  int kk = (int)(i / N), n = (int)(i % N);
  wT[(size_t)n * K + kk] = (bf16_t)w[i];
}

// ---------------------------------------------------------------------------
// LN1 + window partition (zero padding) : x fp32 [B,H,W,C] -> xw bf16 [MQ,C]
// ---------------------------------------------------------------------------
__global__ __launch_bounds__(256) void k_ln1_window(const float* __restrict__ x,
                                                    const float* __restrict__ g,
                                                    const float* __restrict__ b,
                                                    bf16_t* __restrict__ xw) {
  const int token = blockIdx.x;            // 0..MQ-1
  const int win = token / NT, pos = token % NT;
  const int bb = win / (NWIN * NWIN);
  const int ih = (win % (NWIN * NWIN)) / NWIN, iw = win % NWIN;
  const int h = ih * WS_ + pos / WS_;
  const int w = iw * WS_ + pos % WS_;
  bf16_t* outp = xw + (size_t)token * CC;
  if (h >= HH || w >= WW) {
    for (int c = threadIdx.x; c < CC; c += 256) outp[c] = (bf16_t)0.0f;
    return;
  }
  const float* xp = x + (((size_t)bb * HH + h) * WW + w) * CC;
  float vals[3], s = 0.f, s2 = 0.f;
#pragma unroll
  for (int i = 0; i < 3; ++i) {
    float v = xp[threadIdx.x + i * 256];
    vals[i] = v; s += v; s2 += v * v;
  }
  for (int off = 16; off > 0; off >>= 1) { s += __shfl_xor(s, off, 32); s2 += __shfl_xor(s2, off, 32); }
  __shared__ float ps[8], ps2[8], mv[2];
  if ((threadIdx.x & 31) == 0) { ps[threadIdx.x >> 5] = s; ps2[threadIdx.x >> 5] = s2; }
  __syncthreads();
  if (threadIdx.x == 0) {
    float t = 0.f, t2 = 0.f;
#pragma unroll
    for (int i = 0; i < 8; ++i) { t += ps[i]; t2 += ps2[i]; }
    float mean = t / CC;
    mv[0] = mean; mv[1] = t2 / CC - mean * mean;
  }
  __syncthreads();
  const float mean = mv[0], rstd = rsqrtf(mv[1] + 1e-6f);
#pragma unroll
  for (int i = 0; i < 3; ++i) {
    int c = threadIdx.x + i * 256;
    outp[c] = (bf16_t)((vals[i] - mean) * rstd * g[c] + b[c]);
  }
}

// LN over fp32 rows [rows, C] -> bf16 rows
__global__ __launch_bounds__(256) void k_ln_rows(const float* __restrict__ x,
                                                 const float* __restrict__ g,
                                                 const float* __restrict__ b,
                                                 bf16_t* __restrict__ y) {
  const float* xp = x + (size_t)blockIdx.x * CC;
  bf16_t* outp = y + (size_t)blockIdx.x * CC;
  float vals[3], s = 0.f, s2 = 0.f;
#pragma unroll
  for (int i = 0; i < 3; ++i) {
    float v = xp[threadIdx.x + i * 256];
    vals[i] = v; s += v; s2 += v * v;
  }
  for (int off = 16; off > 0; off >>= 1) { s += __shfl_xor(s, off, 32); s2 += __shfl_xor(s2, off, 32); }
  __shared__ float ps[8], ps2[8], mv[2];
  if ((threadIdx.x & 31) == 0) { ps[threadIdx.x >> 5] = s; ps2[threadIdx.x >> 5] = s2; }
  __syncthreads();
  if (threadIdx.x == 0) {
    float t = 0.f, t2 = 0.f;
#pragma unroll
    for (int i = 0; i < 8; ++i) { t += ps[i]; t2 += ps2[i]; }
    float mean = t / CC;
    mv[0] = mean; mv[1] = t2 / CC - mean * mean;
  }
  __syncthreads();
  const float mean = mv[0], rstd = rsqrtf(mv[1] + 1e-6f);
#pragma unroll
  for (int i = 0; i < 3; ++i) {
    int c = threadIdx.x + i * 256;
    outp[c] = (bf16_t)((vals[i] - mean) * rstd * g[c] + b[c]);
  }
}

// ---------------------------------------------------------------------------
// Generic bf16 WMMA GEMM: C = A[M,K] @ Bt[N,K]^T + bias, templated epilogue.
// Block: 256 thr = 8 waves; tile 128(M) x 64(N); each wave: 16 x 64.
// ---------------------------------------------------------------------------
#define MODE_QKV  0
#define MODE_PROJ 1
#define MODE_FC1  2
#define MODE_FC2  3

template <int MODE>
__global__ __launch_bounds__(256) void k_gemm(
    const bf16_t* __restrict__ A, const bf16_t* __restrict__ Bt,
    const float* __restrict__ bias, int M, int N, int K,
    bf16_t* __restrict__ qo, bf16_t* __restrict__ ko, bf16_t* __restrict__ vTo,
    float* __restrict__ out_f32, const float* __restrict__ residual,
    bf16_t* __restrict__ out_bf16) {
  const int lane = threadIdx.x & 31, wave = threadIdx.x >> 5;
  const int l15 = lane & 15, lh = lane >> 4;
  const int mBase = blockIdx.x * 128 + wave * 16;
  const int nBase = blockIdx.y * 64;

  int arow = mBase + l15; if (arow >= M) arow = M - 1;
  const bf16_t* aptr = A + (size_t)arow * K + lh * 16;
  const bf16_t* bptr = Bt + (size_t)(nBase + l15) * K + lh * 16;

  v8f acc[4] = {};
  for (int k = 0; k < K; k += 32) {
    v16bf a = load_a16(aptr + k);
#pragma unroll
    for (int t = 0; t < 4; ++t) {
      v16bf bmat = load_contig16(bptr + (size_t)t * 16 * K + k);
      acc[t] = wmma_bf16(a, bmat, acc[t]);
    }
  }

#pragma unroll
  for (int t = 0; t < 4; ++t) {
#pragma unroll
    for (int r = 0; r < 8; ++r) {
      const int m = mBase + r + lh * 8;
      const int n = nBase + t * 16 + l15;
      if (m >= M) continue;
      float v = acc[t][r] + bias[n];
      if (MODE == MODE_QKV) {
        const int which = n / CC, head = (n % CC) / HD_, hd = n % HD_;
        const int pair = (m / NT) * NH_ + head, pos = m % NT;
        if (which == 0)      qo[((size_t)pair * NT + pos) * HD_ + hd] = (bf16_t)v;
        else if (which == 1) ko[((size_t)pair * NT + pos) * HD_ + hd] = (bf16_t)v;
        else                 vTo[((size_t)pair * HD_ + hd) * VPAD + pos] = (bf16_t)v;
      } else if (MODE == MODE_PROJ) {
        const int win = m / NT, pos = m % NT;
        const int bb = win / (NWIN * NWIN);
        const int ih = (win % (NWIN * NWIN)) / NWIN, iw = win % NWIN;
        const int h = ih * WS_ + pos / WS_, w = iw * WS_ + pos % WS_;
        if (h < HH && w < WW) {
          const size_t o = (((size_t)bb * HH + h) * WW + w) * CC + n;
          out_f32[o] = v + residual[o];
        }
      } else if (MODE == MODE_FC1) {
        const float gelu = 0.5f * v * (1.0f + erff(v * 0.70710678118654752f));
        out_bf16[(size_t)m * N + n] = (bf16_t)gelu;
      } else {  // MODE_FC2
        const size_t o = (size_t)m * N + n;
        out_f32[o] = v + residual[o];
      }
    }
  }
}

// ---------------------------------------------------------------------------
// Window attention, online-softmax (flash) version to keep VGPR pressure low:
// key tiles processed in two halves (8 tiles, then 5 tiles) with running
// row-max / row-sum rescaling of the output accumulators.
// One block per (window, head); 256 thr = 8 waves; each wave owns query tiles.
// ---------------------------------------------------------------------------
template <int TBASE, int NTL>
static __device__ __forceinline__ void attn_half(
    const bf16_t* __restrict__ qb, const bf16_t* __restrict__ kb,
    const bf16_t* __restrict__ vb, const float* __restrict__ relh_s,
    const float* __restrict__ relw_s, bf16_t* __restrict__ myst,
    int mt, int l15, int lh, float* mrun, float* srun, v8f* o) {
  int arow = mt * 16 + l15; if (arow > NT - 1) arow = NT - 1;
  const bf16_t* aptr = qb + (size_t)arow * HD_ + lh * 16;

  // S chunk = q @ k^T for key tiles TBASE .. TBASE+NTL-1
  v8f acc[NTL] = {};
#pragma unroll
  for (int s = 0; s < 2; ++s) {
    v16bf a = load_a16(aptr + s * 32);
#pragma unroll
    for (int tt = 0; tt < NTL; ++tt) {
      int krow = (TBASE + tt) * 16 + l15; if (krow > NT - 1) krow = NT - 1;
      v16bf bmat = load_contig16(kb + (size_t)krow * HD_ + s * 32 + lh * 16);
      acc[tt] = wmma_bf16(a, bmat, acc[tt]);
    }
  }

  // scale + decomposed rel-pos bias + key-padding mask
  const float scale = 0.125f;  // 64^-0.5
#pragma unroll
  for (int tt = 0; tt < NTL; ++tt) {
#pragma unroll
    for (int r = 0; r < 8; ++r) {
      const int m = mt * 16 + r + lh * 8;
      const int n = (TBASE + tt) * 16 + l15;
      float vv;
      if (n > NT - 1) {
        vv = -INFINITY;
      } else {
        const int mm = m > NT - 1 ? NT - 1 : m;
        vv = acc[tt][r] * scale + relh_s[mm * 16 + n / WS_] + relw_s[mm * 16 + n % WS_];
      }
      acc[tt][r] = vv;
    }
  }

  // online softmax update (rows live across 16 lanes of each half-wave)
#pragma unroll
  for (int r = 0; r < 8; ++r) {
    float cmax = -INFINITY;
#pragma unroll
    for (int tt = 0; tt < NTL; ++tt) cmax = fmaxf(cmax, acc[tt][r]);
    for (int off = 8; off > 0; off >>= 1) cmax = fmaxf(cmax, __shfl_xor(cmax, off, 16));
    const float mnew = fmaxf(mrun[r], cmax);
    const float corr = __expf(mrun[r] - mnew);
    float csum = 0.f;
#pragma unroll
    for (int tt = 0; tt < NTL; ++tt) {
      float e = __expf(acc[tt][r] - mnew);
      acc[tt][r] = e; csum += e;
    }
    for (int off = 8; off > 0; off >>= 1) csum += __shfl_xor(csum, off, 16);
    srun[r] = srun[r] * corr + csum;
    mrun[r] = mnew;
#pragma unroll
    for (int nt = 0; nt < 4; ++nt) o[nt][r] *= corr;
  }

  // O += P @ V, restaging P (C-layout -> A-layout) through per-wave LDS
  const int CH = (NTL + 1) / 2;
#pragma unroll
  for (int kk = 0; kk < CH; ++kk) {
    const int t1 = kk * 2 + 1;
#pragma unroll
    for (int r = 0; r < 8; ++r) {
      const int mloc = r + lh * 8;
      myst[mloc * 32 + l15] = (bf16_t)acc[kk * 2][r];
      myst[mloc * 32 + 16 + l15] = (bf16_t)(t1 < NTL ? acc[t1 < NTL ? t1 : 0][r] : 0.0f);
    }
    asm volatile("s_wait_dscnt 0" ::: "memory");
    v16bf pa = load_a16(myst + l15 * 32 + lh * 16);
#pragma unroll
    for (int nt = 0; nt < 4; ++nt) {
      v16bf bmat = load_contig16(vb + (size_t)(nt * 16 + l15) * VPAD +
                                 TBASE * 16 + kk * 32 + lh * 16);
      o[nt] = wmma_bf16(pa, bmat, o[nt]);
    }
  }
}

__global__ __launch_bounds__(256) void k_attn(
    const bf16_t* __restrict__ q, const bf16_t* __restrict__ k,
    const bf16_t* __restrict__ vT, const float* __restrict__ rel_pos_h,
    const float* __restrict__ rel_pos_w, bf16_t* __restrict__ attn_out) {
  __shared__ float relh_s[NT * 16];
  __shared__ float relw_s[NT * 16];
  __shared__ __align__(16) bf16_t pstage[8 * 16 * 32];

  const int pair = blockIdx.x;
  const int win = pair / NH_, head = pair % NH_;
  const bf16_t* qb = q + (size_t)pair * NT * HD_;
  const bf16_t* kb = k + (size_t)pair * NT * HD_;
  const bf16_t* vb = vT + (size_t)pair * HD_ * VPAD;

  // decomposed relative-position bias rows: rel[m][j] = q[m,:] . rel_pos[idx(m,j),:]
  for (int idx = threadIdx.x; idx < NT * WS_ * 2; idx += 256) {
    const int table = idx / (NT * WS_);
    const int rem = idx % (NT * WS_);
    const int m = rem / WS_, j = rem % WS_;
    const float* rp = (table == 0)
        ? rel_pos_h + ((size_t)((m / WS_) - j + (WS_ - 1))) * HD_
        : rel_pos_w + ((size_t)((m % WS_) - j + (WS_ - 1))) * HD_;
    const bf16_t* qr = qb + (size_t)m * HD_;
    float s = 0.f;
#pragma unroll 8
    for (int c = 0; c < HD_; ++c) s += (float)qr[c] * rp[c];
    (table == 0 ? relh_s : relw_s)[m * 16 + j] = s;
  }
  __syncthreads();

  const int lane = threadIdx.x & 31, wave = threadIdx.x >> 5;
  const int l15 = lane & 15, lh = lane >> 4;
  bf16_t* myst = pstage + wave * 512;

  for (int mt = wave; mt < 13; mt += 8) {   // 13 query tiles of 16 (196 -> 208)
    float mrun[8], srun[8];
#pragma unroll
    for (int r = 0; r < 8; ++r) { mrun[r] = -INFINITY; srun[r] = 0.f; }
    v8f o[4] = {};

    // key tiles 0..7 (keys 0..127), then 8..12 (keys 128..207, +pad to 224)
    attn_half<0, 8>(qb, kb, vb, relh_s, relw_s, myst, mt, l15, lh, mrun, srun, o);
    attn_half<8, 5>(qb, kb, vb, relh_s, relw_s, myst, mt, l15, lh, mrun, srun, o);

#pragma unroll
    for (int nt = 0; nt < 4; ++nt) {
#pragma unroll
      for (int r = 0; r < 8; ++r) {
        const int m = mt * 16 + r + lh * 8;
        if (m > NT - 1) continue;
        const float outv = o[nt][r] / srun[r];
        attn_out[((size_t)win * NT + m) * CC + head * HD_ + nt * 16 + l15] = (bf16_t)outv;
      }
    }
  }
}

// ---------------------------------------------------------------------------
// Host launch
// ---------------------------------------------------------------------------
static inline size_t alup(size_t x) { return (x + 255) & ~(size_t)255; }

extern "C" void kernel_launch(void* const* d_in, const int* in_sizes, int n_in,
                              void* d_out, int out_size, void* d_ws, size_t ws_size,
                              hipStream_t stream) {
  (void)in_sizes; (void)n_in; (void)out_size; (void)ws_size;
  const float* x        = (const float*)d_in[0];
  const float* ln1_g    = (const float*)d_in[1];
  const float* ln1_b    = (const float*)d_in[2];
  const float* w_qkv    = (const float*)d_in[3];
  const float* b_qkv    = (const float*)d_in[4];
  const float* w_proj   = (const float*)d_in[5];
  const float* b_proj   = (const float*)d_in[6];
  const float* rel_h    = (const float*)d_in[7];
  const float* rel_w    = (const float*)d_in[8];
  const float* ln2_g    = (const float*)d_in[9];
  const float* ln2_b    = (const float*)d_in[10];
  const float* w_fc1    = (const float*)d_in[11];
  const float* b_fc1    = (const float*)d_in[12];
  const float* w_fc2    = (const float*)d_in[13];
  const float* b_fc2    = (const float*)d_in[14];
  float* out            = (float*)d_out;

  char* ws = (char*)d_ws;
  const size_t SZ_WT_QKV  = (size_t)3 * CC * CC * 2;
  const size_t SZ_WT_PROJ = (size_t)CC * CC * 2;
  const size_t SZ_WT_FC1  = (size_t)4 * CC * CC * 2;
  const size_t SZ_WT_FC2  = (size_t)4 * CC * CC * 2;
  const size_t SZ_XW      = (size_t)MQ * CC * 2;
  const size_t SZ_QK      = (size_t)NPAIR * NT * HD_ * 2;
  const size_t SZ_VT      = (size_t)NPAIR * HD_ * VPAD * 2;
  const size_t SZ_AO      = SZ_XW;
  const size_t SZ_X2      = (size_t)MT * CC * 4;

  size_t off = 0;
  const size_t o_wqkv  = off; off = alup(off + SZ_WT_QKV);
  const size_t o_wproj = off; off = alup(off + SZ_WT_PROJ);
  const size_t o_wfc1  = off; off = alup(off + SZ_WT_FC1);
  const size_t o_wfc2  = off; off = alup(off + SZ_WT_FC2);
  const size_t o_xw    = off; off = alup(off + SZ_XW);
  const size_t o_q     = off; off = alup(off + SZ_QK);
  const size_t o_k     = off; off = alup(off + SZ_QK);
  const size_t o_vt    = off; off = alup(off + SZ_VT);
  const size_t o_ao    = off; off = alup(off + SZ_AO);
  const size_t o_x2    = off; /* end */
  const size_t o_h     = o_xw;   // reuse xw+q+k+vT (dead by FC1 time)
  const size_t o_y     = o_ao;   // reuse attn_out (dead by LN2 time)

  bf16_t* wT_qkv  = (bf16_t*)(ws + o_wqkv);
  bf16_t* wT_proj = (bf16_t*)(ws + o_wproj);
  bf16_t* wT_fc1  = (bf16_t*)(ws + o_wfc1);
  bf16_t* wT_fc2  = (bf16_t*)(ws + o_wfc2);
  bf16_t* xw      = (bf16_t*)(ws + o_xw);
  bf16_t* qbuf    = (bf16_t*)(ws + o_q);
  bf16_t* kbuf    = (bf16_t*)(ws + o_k);
  bf16_t* vTbuf   = (bf16_t*)(ws + o_vt);
  bf16_t* ao      = (bf16_t*)(ws + o_ao);
  float*  x2      = (float*)(ws + o_x2);
  bf16_t* hbuf    = (bf16_t*)(ws + o_h);
  bf16_t* ybuf    = (bf16_t*)(ws + o_y);

  // 1) weight transposes (fp32 [K,N] -> bf16 [N,K])
  {
    size_t n;
    n = (size_t)CC * 3 * CC; k_transpose_w<<<(n + 255) / 256, 256, 0, stream>>>(w_qkv, wT_qkv, CC, 3 * CC);
    n = (size_t)CC * CC;     k_transpose_w<<<(n + 255) / 256, 256, 0, stream>>>(w_proj, wT_proj, CC, CC);
    n = (size_t)CC * 4 * CC; k_transpose_w<<<(n + 255) / 256, 256, 0, stream>>>(w_fc1, wT_fc1, CC, 4 * CC);
    n = (size_t)4 * CC * CC; k_transpose_w<<<(n + 255) / 256, 256, 0, stream>>>(w_fc2, wT_fc2, 4 * CC, CC);
  }

  // 2) LN1 + window partition
  k_ln1_window<<<MQ, 256, 0, stream>>>(x, ln1_g, ln1_b, xw);

  // 3) zero V^T (key padding must be exactly 0 for P@V)
  hipMemsetAsync(vTbuf, 0, SZ_VT, stream);

  // 4) QKV GEMM: [19600,768] x [768,2304]
  k_gemm<MODE_QKV><<<dim3((MQ + 127) / 128, (3 * CC) / 64), 256, 0, stream>>>(
      xw, wT_qkv, b_qkv, MQ, 3 * CC, CC, qbuf, kbuf, vTbuf, nullptr, nullptr, nullptr);

  // 5) attention per (window, head)
  k_attn<<<NPAIR, 256, 0, stream>>>(qbuf, kbuf, vTbuf, rel_h, rel_w, ao);

  // 6) proj GEMM + window reverse + residual -> x2
  k_gemm<MODE_PROJ><<<dim3((MQ + 127) / 128, CC / 64), 256, 0, stream>>>(
      ao, wT_proj, b_proj, MQ, CC, CC, nullptr, nullptr, nullptr, x2, x, nullptr);

  // 7) LN2
  k_ln_rows<<<MT, 256, 0, stream>>>(x2, ln2_g, ln2_b, ybuf);

  // 8) FC1 + exact GELU
  k_gemm<MODE_FC1><<<dim3(MT / 128, (4 * CC) / 64), 256, 0, stream>>>(
      ybuf, wT_fc1, b_fc1, MT, 4 * CC, CC, nullptr, nullptr, nullptr, nullptr, nullptr, hbuf);

  // 9) FC2 + residual -> output
  k_gemm<MODE_FC2><<<dim3(MT / 128, CC / 64), 256, 0, stream>>>(
      hbuf, wT_fc2, b_fc2, MT, CC, 4 * CC, nullptr, nullptr, nullptr, out, x2, nullptr);
}
